// GKTMultiHeadAttention_83021717831966
// MI455X (gfx1250) — matrix operand order repaired and checked by reading, not compiled, not run on hardware
//
#include <hip/hip_runtime.h>

// GKT multi-head attention, restructured:
//   M = Wq^T Wk  (tiny GEMM, device-side, fp32 -> bf16 A-fragment layout)
//   att[n,p] = q_p . (M k_p)       -> one bf16 WMMA GEMM + fused fp32 dot
//   probs   = softmax over n (9)
//   vbar[c,p] = sum_n probs * v    -> elementwise fp32, bf16 staged in LDS
//   out = Wv vbar                  -> small bf16 WMMA GEMM, fp32 out
#define NVIEWS 9
#define CCH 256
#define HWPIX 16384            // 128*128
#define PITCH 272              // bf16 pitch for LDS pixel-major tiles (pad vs 64-bank LDS)

typedef __attribute__((ext_vector_type(16))) __bf16 v16bf;
typedef __attribute__((ext_vector_type(8)))  float  v8f;

__device__ inline unsigned short f32_bf16(float f) {
    union { float f; unsigned u; } x; x.f = f;
    unsigned r = x.u + 0x7FFFu + ((x.u >> 16) & 1u);   // round-to-nearest-even
    return (unsigned short)(r >> 16);
}

// ---------------------------------------------------------------------------
// Kernel 1: build M = Wq^T Wk and reformat M, Wv into WMMA A-fragment layout.
// A-fragment (16x32 bf16, wave32): lane L -> row = L%16, g = L/16;
// vector element e -> VGPR v=e/2, half h=e%2, K = 16*g + e (contiguous k order, same as B).
// Fragment storage: [frag = ctile*8+kchunk][lane][e], contiguous 32B per lane.
// ---------------------------------------------------------------------------
__global__ void prep_weights_kernel(const float* __restrict__ Wq,
                                    const float* __restrict__ Wk,
                                    const float* __restrict__ Wv,
                                    unsigned short* __restrict__ Mlay,
                                    unsigned short* __restrict__ Vlay) {
    int bx = blockIdx.x;
    int t  = (bx & 255) * 256 + threadIdx.x;     // 0..65535 element id
    int frag = t >> 9;
    int lane = (t >> 4) & 31;
    int idx  = t & 15;
    int ctile = frag >> 3, kc = frag & 7;
    int row = lane & 15, g = lane >> 4;
    int v = idx >> 1, h = idx & 1;
    int K = 16 * g + idx;
    if (bx < 256) {
        // M[c][c'] = sum_o Wq[o][c] * Wk[o][c']   (fp32 accumulation)
        int c  = ctile * 16 + row;       // A row (output channel of M*k)
        int cp = kc * 32 + K;            // A col (K dim)
        float s = 0.f;
        for (int o = 0; o < CCH; ++o)
            s = fmaf(Wq[o * CCH + c], Wk[o * CCH + cp], s);
        Mlay[t] = f32_bf16(s);
    } else {
        int o = ctile * 16 + row;        // output channel
        int c = kc * 32 + K;             // input channel (K dim)
        Vlay[t] = f32_bf16(Wv[o * CCH + c]);
    }
}

// ---------------------------------------------------------------------------
// Kernel 2: att logits.  Block = 128 thr (4 waves), 64-pixel tile, one view.
// Stage k tile pixel-major bf16 in LDS; per wave: 16 c-tiles x 8 WMMA(K=32),
// then fused fp32 dot with q via D-tile coordinates, shfl-reduce halves.
// ---------------------------------------------------------------------------
__global__ void attn_logits_kernel(const float* __restrict__ q,
                                   const float* __restrict__ k,
                                   const unsigned short* __restrict__ Mlay,
                                   float* __restrict__ att) {
    __shared__ __align__(32) unsigned short klds[64 * PITCH];
    int n  = blockIdx.x >> 8;                 // view 0..8
    int p0 = (blockIdx.x & 255) * 64;         // pixel tile base
    const float* kbase = k + (size_t)n * CCH * HWPIX;

    // stage: global [c][p] fp32 -> LDS [p][c] bf16 (coalesced reads)
    for (int i = threadIdx.x; i < CCH * 64; i += 128) {
        int c = i >> 6, pp = i & 63;
        klds[pp * PITCH + c] = f32_bf16(kbase[(size_t)c * HWPIX + p0 + pp]);
    }
    __syncthreads();

    int lane = threadIdx.x & 31;
    int w    = threadIdx.x >> 5;              // wave -> 16-pixel sub-tile
    int g    = lane >> 4;
    int p    = p0 + w * 16 + (lane & 15);
    const float* qcol = q + (size_t)n * CCH * HWPIX + p;
    const unsigned short* bsrc = &klds[(w * 16 + (lane & 15)) * PITCH + g * 16];

    float attacc = 0.f;
    for (int ct = 0; ct < 16; ++ct) {
        v8f acc = {};
        const v16bf* Af = (const v16bf*)Mlay + ct * 256 + lane;  // frag stride 32 v16bf
#pragma unroll
        for (int kc = 0; kc < 8; ++kc) {
            v16bf a = Af[kc * 32];
            v16bf b = *(const v16bf*)(bsrc + kc * 32);
            acc = __builtin_amdgcn_wmma_f32_16x16x32_bf16(
                false, a, false, b, (short)0, acc, false, false);
        }
        // D layout: VGPR j holds (c = ct*16 + j + 8g, pixel p); dot with q in fp32
#pragma unroll
        for (int j = 0; j < 8; ++j) {
            int c = ct * 16 + j + 8 * g;
            attacc = fmaf(qcol[(size_t)c * HWPIX], acc[j], attacc);
        }
    }
    attacc += __shfl_xor(attacc, 16, 32);     // combine lane-half c-ranges
    if (lane < 16) att[n * HWPIX + p] = attacc;
}

// ---------------------------------------------------------------------------
// Kernel 3: 9-way softmax over views, per pixel, in place.
// ---------------------------------------------------------------------------
__global__ void softmax9_kernel(float* __restrict__ att) {
    int p = blockIdx.x * 256 + threadIdx.x;
    float a[NVIEWS];
    float m = -3.0e38f;
    for (int nn = 0; nn < NVIEWS; ++nn) { a[nn] = att[nn * HWPIX + p]; m = fmaxf(m, a[nn]); }
    float s = 0.f;
    for (int nn = 0; nn < NVIEWS; ++nn) { a[nn] = __expf(a[nn] - m); s += a[nn]; }
    float inv = __frcp_rn(s);
    for (int nn = 0; nn < NVIEWS; ++nn) att[nn * HWPIX + p] = a[nn] * inv;
}

// ---------------------------------------------------------------------------
// Kernel 4: vbar = sum_n probs*v (fp32, streaming all of v once), staged bf16
// pixel-major in LDS, then out = Wv * vbar via WMMA, fp32 stores.
// 256 threads = 8 waves: wave w -> pixel subtile (w&3), otile half (w>>2).
// Output tiles are independent across waves, so no cross-wave reduction.
// ---------------------------------------------------------------------------
__global__ void out_gemm_kernel(const float* __restrict__ v,
                                const float* __restrict__ probs,
                                const unsigned short* __restrict__ Vlay,
                                float* __restrict__ out) {
    __shared__ __align__(32) unsigned short vlds[64 * PITCH];
    __shared__ float slds[NVIEWS * 64];
    int p0 = blockIdx.x * 64;

    for (int i = threadIdx.x; i < NVIEWS * 64; i += 256)
        slds[i] = probs[(i >> 6) * HWPIX + p0 + (i & 63)];
    __syncthreads();

    for (int i = threadIdx.x; i < CCH * 64; i += 256) {
        int c = i >> 6, pp = i & 63;
        float acc = 0.f;
#pragma unroll
        for (int nn = 0; nn < NVIEWS; ++nn)
            acc = fmaf(slds[nn * 64 + pp],
                       v[(size_t)nn * CCH * HWPIX + (size_t)c * HWPIX + p0 + pp], acc);
        vlds[pp * PITCH + c] = f32_bf16(acc);
    }
    __syncthreads();

    int lane = threadIdx.x & 31;
    int w    = threadIdx.x >> 5;              // 0..7
    int sub  = w & 3;                         // 16-pixel subtile
    int half = w >> 2;                        // otile half: 0 -> 0..7, 1 -> 8..15
    int g    = lane >> 4;
    int p    = p0 + sub * 16 + (lane & 15);
    const unsigned short* bsrc = &vlds[(sub * 16 + (lane & 15)) * PITCH + g * 16];

    for (int ot = half * 8; ot < half * 8 + 8; ++ot) {
        v8f acc = {};
        const v16bf* Af = (const v16bf*)Vlay + ot * 256 + lane;
#pragma unroll
        for (int kc = 0; kc < 8; ++kc) {
            v16bf a = Af[kc * 32];
            v16bf b = *(const v16bf*)(bsrc + kc * 32);
            acc = __builtin_amdgcn_wmma_f32_16x16x32_bf16(
                false, a, false, b, (short)0, acc, false, false);
        }
#pragma unroll
        for (int j = 0; j < 8; ++j)
            out[(size_t)(ot * 16 + j + 8 * g) * HWPIX + p] = acc[j];
    }
}

// ---------------------------------------------------------------------------
extern "C" void kernel_launch(void* const* d_in, const int* in_sizes, int n_in,
                              void* d_out, int out_size, void* d_ws, size_t ws_size,
                              hipStream_t stream) {
    (void)in_sizes; (void)n_in; (void)out_size; (void)ws_size;
    const float* q  = (const float*)d_in[0];
    const float* k  = (const float*)d_in[1];
    const float* v  = (const float*)d_in[2];
    const float* Wq = (const float*)d_in[3];
    const float* Wk = (const float*)d_in[4];
    const float* Wv = (const float*)d_in[5];
    float* out = (float*)d_out;

    char* ws = (char*)d_ws;
    unsigned short* Mlay = (unsigned short*)ws;                 // 128 KB
    unsigned short* Vlay = (unsigned short*)(ws + 131072);      // 128 KB
    float*          att  = (float*)(ws + 262144);               // 9*16384*4 = 576 KB

    prep_weights_kernel<<<512, 256, 0, stream>>>(Wq, Wk, Wv, Mlay, Vlay);
    attn_logits_kernel<<<NVIEWS * 256, 128, 0, stream>>>(q, k, Mlay, att);
    softmax9_kernel<<<HWPIX / 256, 256, 0, stream>>>(att);
    out_gemm_kernel<<<256, 256, 0, stream>>>(v, att, Vlay, out);
}
